// GPT_model_48404281426235
// MI455X (gfx1250) — matrix operand order repaired and checked by reading, not compiled, not run on hardware
//
#include <hip/hip_runtime.h>
#include <hip/hip_bf16.h>
#include <math.h>

#define BB 2
#define SS 1024
#define HH 768
#define NHH 12
#define HD 64
#define LL 4
#define VV 32000
#define MM (BB * SS)   // 2048 token rows

typedef __attribute__((ext_vector_type(16))) __bf16 v16bf;
typedef __attribute__((ext_vector_type(8)))  float  v8f;
typedef unsigned int u32x4 __attribute__((ext_vector_type(4)));
typedef int          i32x8 __attribute__((ext_vector_type(8)));
typedef int          i32x4 __attribute__((ext_vector_type(4)));

// ---------------------------------------------------------------------------
// TDM: 2D tile load Global -> LDS via Tensor Data Mover (ISA ch.8 D# layout).
// group0: [1:0]=count=1, [63:32]=lds_addr, [120:64]=global addr, [127:126]=2
// group1: [17:16]=data_size(2 -> 4B), [79:48]=tensor_dim0, [111:80]=tensor_dim1,
//         [127:112]=tile_dim0, [143:128]=tile_dim1, [207:160]=dim0_stride
// groups 2/3 (and trailing group) unused for a 2D tensor.
// NOTE: this toolchain's builtin is the 6-arg (clang-23 / therock) variant.
// ---------------------------------------------------------------------------
static __device__ inline void tdm_load_2d_f32(unsigned lds_addr, const void* gptr,
                                              int dim0, int dim1, int tile0,
                                              int tile1, int stride0) {
  unsigned long long ga = (unsigned long long)(size_t)gptr;
  u32x4 g0 = { 1u,
               lds_addr,
               (unsigned)ga,
               (unsigned)((ga >> 32) & 0x01ffffffull) | (2u << 30) };
  i32x8 g1 = { (int)(2u << 16),                                      // data_size = 4B
               (int)(((unsigned)dim0 & 0xffffu) << 16),              // dim0 lo
               (int)((((unsigned)dim0 >> 16) & 0xffffu) |
                     (((unsigned)dim1 & 0xffffu) << 16)),            // dim0 hi | dim1 lo
               (int)((((unsigned)dim1 >> 16) & 0xffffu) |
                     (((unsigned)tile0 & 0xffffu) << 16)),           // dim1 hi | tile0
               (int)((unsigned)tile1 & 0xffffu),                     // tile1 (tile2=0)
               stride0,                                              // dim0_stride lo32
               0, 0 };
  i32x4 gz4 = { 0, 0, 0, 0 };
  i32x8 gz8 = { 0, 0, 0, 0, 0, 0, 0, 0 };
  __builtin_amdgcn_tensor_load_to_lds(g0, g1, gz4, gz4, gz8, 0);
}

// ---------------------------------------------------------------------------
// WMMA fragment loaders (CDNA5 16x16x32 bf16 layouts, ISA 7.12.2)
// ---------------------------------------------------------------------------
// A 16x32 (MxK) from bf16 LDS, row-major, leading dim ld.
static __device__ inline v16bf load_a_frag(const __bf16* base, int ld, int lane) {
  int m = lane & 15, kh = lane >> 4;
  const __bf16* row = base + m * ld + 8 * kh;
  v16bf a;
#pragma unroll
  for (int v = 0; v < 8; ++v) {
    int kb = (v < 4) ? (2 * v) : (2 * v + 8);
    a[2 * v]     = row[kb];
    a[2 * v + 1] = row[kb + 1];
  }
  return a;
}

// B 32x16 (KxN) from bf16 LDS stored K-contiguous: element(k,n)=base[n*ld+k].
static __device__ inline v16bf load_bt_frag(const __bf16* base, int ld, int lane) {
  int n = lane & 15, kh = lane >> 4;
  const __bf16* col = base + n * ld + 8 * kh;
  v16bf b;
#pragma unroll
  for (int v = 0; v < 8; ++v) {
    int kb = (v < 4) ? (2 * v) : (2 * v + 8);
    b[2 * v]     = col[kb];
    b[2 * v + 1] = col[kb + 1];
  }
  return b;
}

// A 16x32 from fp32 LDS (TDM-staged), converting to bf16 at frag build.
static __device__ inline v16bf load_a_frag_f32(const float* base, int ld, int lane) {
  int m = lane & 15, kh = lane >> 4;
  const float* row = base + m * ld + 8 * kh;
  v16bf a;
#pragma unroll
  for (int v = 0; v < 8; ++v) {
    int kb = (v < 4) ? (2 * v) : (2 * v + 8);
    a[2 * v]     = (__bf16)row[kb];
    a[2 * v + 1] = (__bf16)row[kb + 1];
  }
  return a;
}

// B 32x16 from fp32 LDS, row-major [k][n] with leading dim ldn.
static __device__ inline v16bf load_b_frag_f32(const float* base, int ldn, int lane) {
  int n = lane & 15, kh = lane >> 4;
  const float* col = base + (size_t)(8 * kh) * ldn + n;
  v16bf b;
#pragma unroll
  for (int v = 0; v < 8; ++v) {
    int kb = (v < 4) ? (2 * v) : (2 * v + 8);
    b[2 * v]     = (__bf16)col[kb * ldn];
    b[2 * v + 1] = (__bf16)col[(kb + 1) * ldn];
  }
  return b;
}

static __device__ inline v8f wmma_bf16(v16bf a, v16bf b, v8f c) {
  return __builtin_amdgcn_wmma_f32_16x16x32_bf16(false, a, false, b, (short)0, c,
                                                 false, false);
}

// ---------------------------------------------------------------------------
// Embedding: h[b,s,:] = bpe[ids[b,s]] + pos[s]
// ---------------------------------------------------------------------------
__global__ __launch_bounds__(256) void embed_kernel(const int* __restrict__ ids,
                                                    const float* __restrict__ bpe,
                                                    const float* __restrict__ pos,
                                                    float* __restrict__ h) {
  size_t idx = (size_t)blockIdx.x * 256 + threadIdx.x;
  int row = (int)(idx / HH);
  int col = (int)(idx % HH);
  int s = row % SS;
  int tok = ids[row];
  h[idx] = bpe[(size_t)tok * HH + col] + pos[(size_t)s * HH + col];
}

// ---------------------------------------------------------------------------
// Tiled bf16-WMMA GEMM:  C[M,N] = act(A[M,K] * W[K,N] + bias)
// BM=BN=128, BK=32. 256 threads = 8 waves (4x2); each wave owns 32x64.
// Tiles staged fp32 via TDM (tensor_load_to_lds), converted to bf16 at
// fragment build. Next tile prefetched into GL2 while computing.
// act: 0 = none, 1 = exact GELU
// ---------------------------------------------------------------------------
__global__ __launch_bounds__(256) void gemm_kernel(const float* __restrict__ A,
                                                   const float* __restrict__ W,
                                                   const float* __restrict__ bias,
                                                   float* __restrict__ C,
                                                   int M, int N, int K, int act) {
  __shared__ float AsF[128][32];   // [m][k]
  __shared__ float BsF[32][128];   // [k][n]

  int tid = threadIdx.x;
  int bm = blockIdx.y * 128, bn = blockIdx.x * 128;
  int wave = tid >> 5, lane = tid & 31;
  int m0 = (wave >> 1) * 32, n0 = (wave & 1) * 64;

  unsigned lds_a = (unsigned)(size_t)(&AsF[0][0]);
  unsigned lds_b = (unsigned)(size_t)(&BsF[0][0]);

  v8f z = {};
  v8f acc[2][4];
#pragma unroll
  for (int i = 0; i < 2; ++i)
#pragma unroll
    for (int j = 0; j < 4; ++j) acc[i][j] = z;

  for (int k0 = 0; k0 < K; k0 += 32) {
    // --- stage current tiles via Tensor Data Mover (wave 0 issues the DMA) ---
    if (wave == 0) {
      // A tile: 128 rows x 32 cols out of A[M,K]
      tdm_load_2d_f32(lds_a, A + (size_t)bm * K + k0,
                      /*dim0*/ K, /*dim1*/ 128, /*tile0*/ 32, /*tile1*/ 128,
                      /*stride0*/ K);
      // W tile: 32 rows x 128 cols out of W[K,N]
      tdm_load_2d_f32(lds_b, W + (size_t)k0 * N + bn,
                      /*dim0*/ N, /*dim1*/ 32, /*tile0*/ 128, /*tile1*/ 32,
                      /*stride0*/ N);
      __builtin_amdgcn_s_wait_tensorcnt(0);
    }
    __syncthreads();

    // --- prefetch next K-tile into GL2 while we compute this one ---
    if (k0 + 32 < K) {
      __builtin_prefetch(A + (size_t)(bm + (tid & 127)) * K + k0 + 32, 0, 1);
      __builtin_prefetch(W + (size_t)(k0 + 32 + (tid & 31)) * N + bn +
                             ((tid >> 5) << 4), 0, 1);
    }

    v16bf af[2], bf[4];
#pragma unroll
    for (int i = 0; i < 2; ++i)
      af[i] = load_a_frag_f32(&AsF[m0 + i * 16][0], 32, lane);
#pragma unroll
    for (int j = 0; j < 4; ++j)
      bf[j] = load_b_frag_f32(&BsF[0][n0 + j * 16], 128, lane);
#pragma unroll
    for (int i = 0; i < 2; ++i)
#pragma unroll
      for (int j = 0; j < 4; ++j) acc[i][j] = wmma_bf16(af[i], bf[j], acc[i][j]);
    __syncthreads();
  }

  int kh = lane >> 4, nl = lane & 15;
  float bv[4];
#pragma unroll
  for (int j = 0; j < 4; ++j)
    bv[j] = bias ? bias[bn + n0 + j * 16 + nl] : 0.0f;

#pragma unroll
  for (int i = 0; i < 2; ++i)
#pragma unroll
    for (int j = 0; j < 4; ++j)
#pragma unroll
      for (int r = 0; r < 8; ++r) {
        int row = bm + m0 + i * 16 + r + 8 * kh;
        int col = bn + n0 + j * 16 + nl;
        float x = acc[i][j][r] + bv[j];
        if (act) x = 0.5f * x * (1.0f + erff(x * 0.70710678118f));
        C[(size_t)row * N + col] = x;
      }
}

// ---------------------------------------------------------------------------
// Flash attention over qkv[B,S,3H]. 128 threads = 4 waves; each wave handles
// 16 query rows with online softmax, streaming 32-key K/V tiles through
// wave-private LDS. No __syncthreads needed (per-wave LDS, in-order DS).
// ---------------------------------------------------------------------------
__global__ __launch_bounds__(128) void attn_kernel(const float* __restrict__ qkv,
                                                   const float* __restrict__ seq_mask,
                                                   float* __restrict__ av) {
  __shared__ __bf16 sK[4][32][66];  // [key][dim]
  __shared__ __bf16 sV[4][64][34];  // transposed: [dim][key]
  __shared__ __bf16 sP[4][16][34];  // probs [q][key]

  int tid = threadIdx.x, wave = tid >> 5, lane = tid & 31;
  const int nqb = SS / 64;
  int bid = blockIdx.x;
  int qblk = bid % nqb;
  int head = (bid / nqb) % NHH;
  int b = bid / (nqb * NHH);
  int q0 = qblk * 64 + wave * 16;
  int m = lane & 15, kh = lane >> 4, nl = m;
  size_t rowbase = (size_t)b * SS;

  // Q fragments, pre-scaled by 1/sqrt(HD) = 0.125
  v16bf aq[2];
  const float* qrow = qkv + (rowbase + q0 + m) * (3 * HH) + head * HD;
#pragma unroll
  for (int c = 0; c < 2; ++c)
#pragma unroll
    for (int v = 0; v < 8; ++v) {
      int kb = c * 32 + 8 * kh + ((v < 4) ? 2 * v : 2 * v + 8);
      aq[c][2 * v]     = (__bf16)(qrow[kb] * 0.125f);
      aq[c][2 * v + 1] = (__bf16)(qrow[kb + 1] * 0.125f);
    }

  v8f zv = {};
  float mrow[8], lrow[8];
  v8f o[4];
#pragma unroll
  for (int r = 0; r < 8; ++r) { mrow[r] = -INFINITY; lrow[r] = 0.0f; }
#pragma unroll
  for (int j = 0; j < 4; ++j) o[j] = zv;

  for (int k0 = 0; k0 <= q0 + 15; k0 += 32) {
    // stage 32-key K and V tiles (bf16; V stored transposed)
    for (int g = lane; g < 512; g += 32) {
      int r = g >> 4, c4 = (g & 15) << 2;
      const float* kp = qkv + (rowbase + k0 + r) * (3 * HH) + HH + head * HD + c4;
      float4 kv4 = *(const float4*)kp;
      sK[wave][r][c4 + 0] = (__bf16)kv4.x;
      sK[wave][r][c4 + 1] = (__bf16)kv4.y;
      sK[wave][r][c4 + 2] = (__bf16)kv4.z;
      sK[wave][r][c4 + 3] = (__bf16)kv4.w;
      float4 vv4 = *(const float4*)(kp + HH);
      sV[wave][c4 + 0][r] = (__bf16)vv4.x;
      sV[wave][c4 + 1][r] = (__bf16)vv4.y;
      sV[wave][c4 + 2][r] = (__bf16)vv4.z;
      sV[wave][c4 + 3][r] = (__bf16)vv4.w;
    }

    // scores: two 16x16 tiles, K-dim = 64 via two chained WMMAs
    v8f s[2];
#pragma unroll
    for (int t = 0; t < 2; ++t) {
      v8f c = zv;
#pragma unroll
      for (int cc = 0; cc < 2; ++cc) {
        v16bf bk = load_bt_frag(&sK[wave][t * 16][cc * 32], 66, lane);
        c = wmma_bf16(aq[cc], bk, c);
      }
      s[t] = c;
    }

    float sm0 = seq_mask[rowbase + k0 + nl];
    float sm1 = seq_mask[rowbase + k0 + 16 + nl];
#pragma unroll
    for (int r = 0; r < 8; ++r) {
      int qi = q0 + r + 8 * kh;
      float e0 = s[0][r] + (1.0f - (((k0 + nl) <= qi) ? sm0 : 0.0f)) * -1e9f;
      float e1 = s[1][r] + (1.0f - (((k0 + 16 + nl) <= qi) ? sm1 : 0.0f)) * -1e9f;
      float rm = fmaxf(e0, e1);
#pragma unroll
      for (int msk = 1; msk < 16; msk <<= 1) rm = fmaxf(rm, __shfl_xor(rm, msk, 32));
      float nm = fmaxf(mrow[r], rm);
      float p0 = __expf(e0 - nm), p1 = __expf(e1 - nm);
      float rs = p0 + p1;
#pragma unroll
      for (int msk = 1; msk < 16; msk <<= 1) rs += __shfl_xor(rs, msk, 32);
      float corr = __expf(mrow[r] - nm);
      lrow[r] = lrow[r] * corr + rs;
      mrow[r] = nm;
#pragma unroll
      for (int j = 0; j < 4; ++j) o[j][r] *= corr;
      int pr = r + 8 * kh;
      sP[wave][pr][nl]      = (__bf16)p0;
      sP[wave][pr][nl + 16] = (__bf16)p1;
    }

    // O += P(16x32) x V(32x64)
    v16bf ap = load_a_frag(&sP[wave][0][0], 34, lane);
#pragma unroll
    for (int j = 0; j < 4; ++j) {
      v16bf bv = load_bt_frag(&sV[wave][j * 16][0], 34, lane);
      o[j] = wmma_bf16(ap, bv, o[j]);
    }
  }

#pragma unroll
  for (int j = 0; j < 4; ++j)
#pragma unroll
    for (int r = 0; r < 8; ++r) {
      int row = q0 + r + 8 * kh;
      av[(rowbase + row) * HH + head * HD + j * 16 + nl] = o[j][r] / lrow[r];
    }
}

// ---------------------------------------------------------------------------
// Block reductions
// ---------------------------------------------------------------------------
static __device__ inline float block_sum(float v, float* red) {
#pragma unroll
  for (int msk = 1; msk < 32; msk <<= 1) v += __shfl_xor(v, msk, 32);
  int wave = threadIdx.x >> 5, lane = threadIdx.x & 31;
  __syncthreads();
  if (lane == 0) red[wave] = v;
  __syncthreads();
  int nw = (blockDim.x + 31) >> 5;
  float t = (lane < nw) ? red[lane] : 0.0f;
#pragma unroll
  for (int msk = 1; msk < 8; msk <<= 1) t += __shfl_xor(t, msk, 32);
  return t;
}

static __device__ inline float block_max(float v, float* red) {
#pragma unroll
  for (int msk = 1; msk < 32; msk <<= 1) v = fmaxf(v, __shfl_xor(v, msk, 32));
  int wave = threadIdx.x >> 5, lane = threadIdx.x & 31;
  __syncthreads();
  if (lane == 0) red[wave] = v;
  __syncthreads();
  int nw = (blockDim.x + 31) >> 5;
  float t = (lane < nw) ? red[lane] : -INFINITY;
#pragma unroll
  for (int msk = 1; msk < 8; msk <<= 1) t = fmaxf(t, __shfl_xor(t, msk, 32));
  return t;
}

// ---------------------------------------------------------------------------
// h = LayerNorm(h + a) ; one block per token row (768 = 3 * 256)
// ---------------------------------------------------------------------------
__global__ __launch_bounds__(256) void add_ln_kernel(float* __restrict__ h,
                                                     const float* __restrict__ a,
                                                     const float* __restrict__ g,
                                                     const float* __restrict__ bta) {
  __shared__ float red[8];
  int row = blockIdx.x, t = threadIdx.x;
  float* hr = h + (size_t)row * HH;
  const float* ar = a + (size_t)row * HH;
  float x[3];
#pragma unroll
  for (int i = 0; i < 3; ++i) x[i] = hr[t + i * 256] + ar[t + i * 256];
  float mean = block_sum(x[0] + x[1] + x[2], red) * (1.0f / HH);
  float vs = 0.0f;
#pragma unroll
  for (int i = 0; i < 3; ++i) { float d = x[i] - mean; vs += d * d; }
  float var = block_sum(vs, red) * (1.0f / HH);
  float inv = rsqrtf(var + 1e-5f);
#pragma unroll
  for (int i = 0; i < 3; ++i) {
    int c = t + i * 256;
    hr[c] = g[c] * ((x[i] - mean) * inv) + bta[c];
  }
}

// ---------------------------------------------------------------------------
// softmax over V per row: prob = softmax(score)
// ---------------------------------------------------------------------------
__global__ __launch_bounds__(256) void softmax_kernel(const float* __restrict__ score,
                                                      float* __restrict__ prob, int N) {
  __shared__ float red[8];
  int row = blockIdx.x;
  const float* s = score + (size_t)row * N;
  float* p = prob + (size_t)row * N;
  float lm = -INFINITY;
  for (int c = threadIdx.x; c < N; c += 256) lm = fmaxf(lm, s[c]);
  float mx = block_max(lm, red);
  float ls = 0.0f;
  for (int c = threadIdx.x; c < N; c += 256) ls += __expf(s[c] - mx);
  float inv = 1.0f / block_sum(ls, red);
  for (int c = threadIdx.x; c < N; c += 256) p[c] = __expf(s[c] - mx) * inv;
}

// ---------------------------------------------------------------------------
// Orchestration
// ---------------------------------------------------------------------------
extern "C" void kernel_launch(void* const* d_in, const int* in_sizes, int n_in,
                              void* d_out, int out_size, void* d_ws, size_t ws_size,
                              hipStream_t stream) {
  const int*   ids      = (const int*)d_in[0];
  const float* seq_mask = (const float*)d_in[1];
  const float* bpe      = (const float*)d_in[2];
  const float* pos      = (const float*)d_in[3];
  const float* qkv_w    = (const float*)d_in[4];
  const float* qkv_b    = (const float*)d_in[5];
  const float* res_w    = (const float*)d_in[6];
  const float* res_b    = (const float*)d_in[7];
  const float* ln1_g    = (const float*)d_in[8];
  const float* ln1_b    = (const float*)d_in[9];
  const float* ff1_w    = (const float*)d_in[10];
  const float* ff1_b    = (const float*)d_in[11];
  const float* ff2_w    = (const float*)d_in[12];
  const float* ff2_b    = (const float*)d_in[13];
  const float* ln2_g    = (const float*)d_in[14];
  const float* ln2_b    = (const float*)d_in[15];
  const float* pred_w   = (const float*)d_in[16];

  float* out = (float*)d_out;
  float* ws = (float*)d_ws;
  float* h     = ws;                         // [M,H]
  float* a     = h + (size_t)MM * HH;        // [M,H]
  float* qkv   = a + (size_t)MM * HH;        // [M,3H]
  float* av    = qkv + (size_t)MM * 3 * HH;  // [M,H]
  float* ffmid = av + (size_t)MM * HH;       // [M,4H]

  embed_kernel<<<(MM * HH) / 256, 256, 0, stream>>>(ids, bpe, pos, h);

  for (int l = 0; l < LL; ++l) {
    gemm_kernel<<<dim3((3 * HH) / 128, MM / 128), 256, 0, stream>>>(
        h, qkv_w + (size_t)l * HH * 3 * HH, qkv_b + (size_t)l * 3 * HH, qkv,
        MM, 3 * HH, HH, 0);
    attn_kernel<<<BB * NHH * (SS / 64), 128, 0, stream>>>(qkv, seq_mask, av);
    gemm_kernel<<<dim3(HH / 128, MM / 128), 256, 0, stream>>>(
        av, res_w + (size_t)l * HH * HH, res_b + (size_t)l * HH, a,
        MM, HH, HH, 0);
    add_ln_kernel<<<MM, 256, 0, stream>>>(h, a, ln1_g + (size_t)l * HH,
                                          ln1_b + (size_t)l * HH);
    gemm_kernel<<<dim3((4 * HH) / 128, MM / 128), 256, 0, stream>>>(
        h, ff1_w + (size_t)l * HH * 4 * HH, ff1_b + (size_t)l * 4 * HH, ffmid,
        MM, 4 * HH, HH, 1);
    gemm_kernel<<<dim3(HH / 128, MM / 128), 256, 0, stream>>>(
        ffmid, ff2_w + (size_t)l * 4 * HH * HH, ff2_b + (size_t)l * HH, a,
        MM, HH, 4 * HH, 0);
    add_ln_kernel<<<MM, 256, 0, stream>>>(h, a, ln2_g + (size_t)l * HH,
                                          ln2_b + (size_t)l * HH);
  }

  // logits straight into d_out, then softmax into the second half
  gemm_kernel<<<dim3(VV / 128, MM / 128), 256, 0, stream>>>(
      h, pred_w, nullptr, out, MM, VV, HH, 0);
  softmax_kernel<<<MM, 256, 0, stream>>>(out, out + (size_t)MM * VV, VV);
}
